// AdaptiveSparseAttention_61048665145796
// MI455X (gfx1250) — matrix operand order repaired and checked by reading, not compiled, not run on hardware
//
#include <hip/hip_runtime.h>

// ---------------------------------------------------------------------------
// AdaptiveSparseAttention for MI455X (gfx1250), wave32 + WMMA bf16.
// Double-buffered GEMM pipelines (32x64 wave tile), hoisted K/V fragment
// loads in attention, exp2f-based RoPE frequency math.
// ---------------------------------------------------------------------------

#define D_MODEL 2048
#define H_TOTAL 32
#define D_HEAD  64
#define H_ACTIVE 4
#define B_SZ 4
#define T_SZ 1024
#define NTOK (B_SZ * T_SZ)      // 4096 tokens
#define N_QKV (3 * D_MODEL)     // 6144

typedef __attribute__((ext_vector_type(16))) __bf16 v16bf;
typedef __attribute__((ext_vector_type(8)))  float  v8f;

union FragBF {
    v16bf v;
    uint4 u[2];
};

__device__ __forceinline__ unsigned short f2bf(float f) {
    unsigned u = __float_as_uint(f);
    unsigned r = u + 0x7FFFu + ((u >> 16) & 1u);   // round-to-nearest-even
    return (unsigned short)(r >> 16);
}

// A-matrix 16x32 bf16 fragment (ISA 7.12.2): lane m = lane&15, half = lane>>4.
// elements 0..7  -> K = k0 + 8*half + e
// elements 8..15 -> K = k0 + 16 + 8*half + (e-8)
__device__ __forceinline__ FragBF load_a_frag(const unsigned short* __restrict__ base,
                                              int ld, int m0, int k0, int lane) {
    int m = m0 + (lane & 15);
    int half = lane >> 4;
    const unsigned short* p = base + (size_t)m * ld + k0 + 8 * half;
    FragBF f;
    f.u[0] = *(const uint4*)p;
    f.u[1] = *(const uint4*)(p + 16);
    return f;
}

// Same layout but sourced from LDS (16x32 row-major ushort tile).
__device__ __forceinline__ FragBF load_a_frag_lds(const unsigned short* p16, int lane) {
    int m = lane & 15;
    int half = lane >> 4;
    const unsigned short* p = p16 + m * 32 + 8 * half;
    FragBF f;
    f.u[0] = *(const uint4*)p;
    f.u[1] = *(const uint4*)(p + 16);
    return f;
}

// B-matrix 32x16 bf16 fragment from B^T stored (N,K) row-major:
// lane n = n0 + (lane&15); elements 0..7 -> K = k0 + 8*half + e, 8..15 -> K = k0 + 16 + 8*half + (e-8).
__device__ __forceinline__ FragBF load_bt_frag(const unsigned short* __restrict__ bt,
                                               int ld, int n0, int k0, int lane) {
    int n = n0 + (lane & 15);
    int half = lane >> 4;
    const unsigned short* p = bt + (size_t)n * ld + k0 + 8 * half;
    FragBF f;
    f.u[0] = *(const uint4*)p;
    f.u[1] = *(const uint4*)(p + 16);
    return f;
}

__device__ __forceinline__ v8f wmma_bf16(const FragBF& a, const FragBF& b, v8f c) {
    return __builtin_amdgcn_wmma_f32_16x16x32_bf16(false, a.v, false, b.v,
                                                   (short)0, c, false, false);
}

// ---------------------------------------------------------------------------
// 1) Router: one wave per token, lane = head. softmax + top-4 gate.
// ---------------------------------------------------------------------------
__global__ __launch_bounds__(256) void router_kernel(const float* __restrict__ x,
                                                     const float* __restrict__ wr,
                                                     float* __restrict__ gate) {
    int lane = threadIdx.x & 31;
    int wave = threadIdx.x >> 5;
    int tok = blockIdx.x * 8 + wave;
    const float* xr = x + (size_t)tok * D_MODEL;
    float acc = 0.f;
    for (int d = 0; d < D_MODEL; d += 4) {
        float4 xv = *(const float4*)(xr + d);
        acc += xv.x * wr[(d + 0) * H_TOTAL + lane];
        acc += xv.y * wr[(d + 1) * H_TOTAL + lane];
        acc += xv.z * wr[(d + 2) * H_TOTAL + lane];
        acc += xv.w * wr[(d + 3) * H_TOTAL + lane];
    }
    float m = acc;
    #pragma unroll
    for (int off = 1; off < 32; off <<= 1) m = fmaxf(m, __shfl_xor(m, off, 32));
    float e = __expf(acc - m);
    float s = e;
    #pragma unroll
    for (int off = 1; off < 32; off <<= 1) s += __shfl_xor(s, off, 32);
    float prob = e / s;

    float p = prob;
    bool keep = false;
    #pragma unroll
    for (int it = 0; it < H_ACTIVE; ++it) {
        float vm = p;
        #pragma unroll
        for (int off = 1; off < 32; off <<= 1) vm = fmaxf(vm, __shfl_xor(vm, off, 32));
        unsigned msk = (unsigned)__ballot(p == vm);
        int sel = __ffs(msk) - 1;
        if (lane == sel) { keep = true; p = -1.f; }
    }
    gate[(size_t)tok * H_TOTAL + lane] = keep ? prob : 0.f;
}

// ---------------------------------------------------------------------------
// 2) Casts: f32 -> bf16 (x), and transposed cast (weights -> (N,K) bf16).
// ---------------------------------------------------------------------------
__global__ __launch_bounds__(256) void cast_bf16_kernel(const float* __restrict__ src,
                                                        unsigned short* __restrict__ dst,
                                                        int n) {
    int idx = (blockIdx.x * blockDim.x + threadIdx.x) * 4;
    if (idx < n) {
        float4 v = *(const float4*)(src + idx);
        uint2 o;
        o.x = (unsigned)f2bf(v.x) | ((unsigned)f2bf(v.y) << 16);
        o.y = (unsigned)f2bf(v.z) | ((unsigned)f2bf(v.w) << 16);
        *(uint2*)(dst + idx) = o;
    }
}

// src (R,C) f32 row-major -> dst (C,R) bf16 row-major
__global__ __launch_bounds__(256) void transpose_cast_kernel(const float* __restrict__ src,
                                                             unsigned short* __restrict__ dst,
                                                             int R, int C) {
    __shared__ float tile[32][33];
    int c0 = blockIdx.x * 32, r0 = blockIdx.y * 32;
    int tx = threadIdx.x & 31, ty = threadIdx.x >> 5;   // ty in 0..7
    #pragma unroll
    for (int i = 0; i < 4; ++i)
        tile[ty + i * 8][tx] = src[(size_t)(r0 + ty + i * 8) * C + c0 + tx];
    __syncthreads();
    #pragma unroll
    for (int i = 0; i < 4; ++i)
        dst[(size_t)(c0 + ty + i * 8) * R + r0 + tx] = f2bf(tile[tx][ty + i * 8]);
}

// ---------------------------------------------------------------------------
// 3) QKV GEMM (4096 x 6144 x 2048) + fused RoPE + head-major scatter.
//    Block: 256 thr = 8 waves; block tile 256(M) x 64(N); wave tile 32 x 64.
//    Double-buffered fragments: next-k loads in flight during 8 WMMAs.
// ---------------------------------------------------------------------------
__global__ __launch_bounds__(256) void qkv_rope_kernel(const unsigned short* __restrict__ xb,
                                                       const unsigned short* __restrict__ wt,
                                                       unsigned short* __restrict__ qb,
                                                       unsigned short* __restrict__ kb,
                                                       unsigned short* __restrict__ vtb) {
    int lane = threadIdx.x & 31;
    int wave = threadIdx.x >> 5;
    int m0 = blockIdx.y * 256 + wave * 32;
    int n0 = blockIdx.x * 64;
    v8f c[2][4] = {};
    FragBF a[2], b[4], an[2], bn[4];
    a[0] = load_a_frag(xb, D_MODEL, m0,      0, lane);
    a[1] = load_a_frag(xb, D_MODEL, m0 + 16, 0, lane);
    #pragma unroll
    for (int j = 0; j < 4; ++j) b[j] = load_bt_frag(wt, D_MODEL, n0 + j * 16, 0, lane);

    for (int k0 = 0; k0 < D_MODEL; k0 += 32) {
        int kn = k0 + 32;
        if (kn < D_MODEL) {
            an[0] = load_a_frag(xb, D_MODEL, m0,      kn, lane);
            an[1] = load_a_frag(xb, D_MODEL, m0 + 16, kn, lane);
            #pragma unroll
            for (int j = 0; j < 4; ++j)
                bn[j] = load_bt_frag(wt, D_MODEL, n0 + j * 16, kn, lane);
        }
        #pragma unroll
        for (int i = 0; i < 2; ++i)
            #pragma unroll
            for (int j = 0; j < 4; ++j)
                c[i][j] = wmma_bf16(a[i], b[j], c[i][j]);
        a[0] = an[0]; a[1] = an[1];
        #pragma unroll
        for (int j = 0; j < 4; ++j) b[j] = bn[j];
    }

    int half = lane >> 4;
    int nloc = lane & 15;
    const float kRopeLg = -0.20762050593046f;   // -log2(10000)/64
    #pragma unroll
    for (int i = 0; i < 2; ++i) {
        #pragma unroll
        for (int j = 0; j < 4; ++j) {
            int ng = n0 + j * 16;            // 16-aligned global column
            int sec = ng / D_MODEL;          // 0=q 1=k 2=v
            int rem = ng % D_MODEL;
            int h = rem / D_HEAD;
            int dbase = rem % D_HEAD;        // 16-aligned
            #pragma unroll
            for (int r = 0; r < 8; ++r) {
                int row = m0 + i * 16 + r + 8 * half; // token index
                int bb = row / T_SZ, t = row % T_SZ;
                int d = dbase + nloc;
                float val = c[i][j][r];
                if (sec == 2) {              // V, transposed (B,H,D,T)
                    vtb[(((size_t)(bb * H_TOTAL + h)) * D_HEAD + d) * T_SZ + t] = f2bf(val);
                } else {                     // Q/K with RoPE, (B,H,T,D)
                    float other = __shfl_xor(val, 1, 32);
                    float freq = exp2f(kRopeLg * (float)(d & ~1));
                    float ang = (float)t * freq;
                    float sn, cs;
                    __sincosf(ang, &sn, &cs);
                    float res = ((d & 1) == 0) ? (val * cs - other * sn)
                                               : (other * sn + val * cs);
                    size_t idx = (((size_t)(bb * H_TOTAL + h)) * T_SZ + t) * D_HEAD + d;
                    if (sec == 0) qb[idx] = f2bf(res);
                    else          kb[idx] = f2bf(res);
                }
            }
        }
    }
}

// ---------------------------------------------------------------------------
// 4) Flash-style causal attention. One wave per (b,h, 16-row q tile).
//    K processed in 32-wide chunks; all K frags hoisted before S WMMAs,
//    V frags issued before the softmax/LDS round trip.
// ---------------------------------------------------------------------------
__global__ __launch_bounds__(32) void attn_kernel(const unsigned short* __restrict__ qb,
                                                  const unsigned short* __restrict__ kb,
                                                  const unsigned short* __restrict__ vtb,
                                                  const float* __restrict__ gate,
                                                  unsigned short* __restrict__ ob) {
    __shared__ __align__(16) unsigned short pbuf[16 * 32];
    int lane = threadIdx.x & 31;
    int half = lane >> 4, nloc = lane & 15;
    int bh = blockIdx.y;                 // 0..127
    int b = bh >> 5, h = bh & 31;
    int qt = blockIdx.x;                 // 0..63
    const unsigned short* Q  = qb  + (size_t)bh * T_SZ * D_HEAD;
    const unsigned short* K  = kb  + (size_t)bh * T_SZ * D_HEAD;
    const unsigned short* Vt = vtb + (size_t)bh * D_HEAD * T_SZ;

    FragBF aq0 = load_a_frag(Q, D_HEAD, qt * 16, 0, lane);
    FragBF aq1 = load_a_frag(Q, D_HEAD, qt * 16, 32, lane);

    float mrow[8], lrow[8];
    v8f o[4] = {};
    #pragma unroll
    for (int r = 0; r < 8; ++r) { mrow[r] = -1e30f; lrow[r] = 0.f; }

    const float scale = 0.125f;          // 1/sqrt(64)
    int nk32 = (qt * 16 + 15 + 32) / 32; // causal chunk count
    for (int kc = 0; kc < nk32; ++kc) {
        int kbase = kc * 32;
        // hoist all 4 K fragments, then 4 WMMAs in 2 chains of 2
        FragBF bk[2][2];
        #pragma unroll
        for (int sfi = 0; sfi < 2; ++sfi) {
            bk[sfi][0] = load_bt_frag(K, D_HEAD, kbase + sfi * 16, 0, lane);
            bk[sfi][1] = load_bt_frag(K, D_HEAD, kbase + sfi * 16, 32, lane);
        }
        v8f s[2];
        #pragma unroll
        for (int sfi = 0; sfi < 2; ++sfi) {
            v8f acc = {};
            acc = wmma_bf16(aq0, bk[sfi][0], acc);
            acc = wmma_bf16(aq1, bk[sfi][1], acc);
            s[sfi] = acc;
        }
        // issue V loads now so they fly during softmax VALU work
        FragBF bv[4];
        #pragma unroll
        for (int j = 0; j < 4; ++j)
            bv[j] = load_bt_frag(Vt, T_SZ, j * 16, kbase, lane);

        float mt[8];
        #pragma unroll
        for (int r = 0; r < 8; ++r) {
            int qrow = qt * 16 + r + 8 * half;
            float v0 = s[0][r] * scale, v1 = s[1][r] * scale;
            v0 = (kbase + nloc      <= qrow) ? v0 : -1e30f;
            v1 = (kbase + 16 + nloc <= qrow) ? v1 : -1e30f;
            s[0][r] = v0; s[1][r] = v1;
            float m = fmaxf(v0, v1);
            #pragma unroll
            for (int off = 1; off < 16; off <<= 1) m = fmaxf(m, __shfl_xor(m, off, 32));
            mt[r] = m;
        }
        float alpha[8];
        #pragma unroll
        for (int r = 0; r < 8; ++r) {
            float mn = fmaxf(mrow[r], mt[r]);
            alpha[r] = __expf(mrow[r] - mn);
            mrow[r] = mn;
            float p0 = __expf(s[0][r] - mn);
            float p1 = __expf(s[1][r] - mn);
            s[0][r] = p0; s[1][r] = p1;
            float rs = p0 + p1;
            #pragma unroll
            for (int off = 1; off < 16; off <<= 1) rs += __shfl_xor(rs, off, 32);
            lrow[r] = lrow[r] * alpha[r] + rs;
        }
        // C-layout P -> LDS row-major 16x32 -> A-layout fragment
        #pragma unroll
        for (int r = 0; r < 8; ++r) {
            int row = r + 8 * half;
            pbuf[row * 32 + nloc]      = f2bf(s[0][r]);
            pbuf[row * 32 + 16 + nloc] = f2bf(s[1][r]);
        }
        __syncthreads();
        FragBF ap = load_a_frag_lds(pbuf, lane);
        __syncthreads();
        #pragma unroll
        for (int j = 0; j < 4; ++j) {
            #pragma unroll
            for (int r = 0; r < 8; ++r) o[j][r] *= alpha[r];
            o[j] = wmma_bf16(ap, bv[j], o[j]);
        }
    }
    float g[8];
    #pragma unroll
    for (int r = 0; r < 8; ++r) {
        int t = qt * 16 + r + 8 * half;
        g[r] = gate[((size_t)b * T_SZ + t) * H_TOTAL + h] / lrow[r];
    }
    #pragma unroll
    for (int j = 0; j < 4; ++j) {
        #pragma unroll
        for (int r = 0; r < 8; ++r) {
            int t = qt * 16 + r + 8 * half;
            int d = j * 16 + nloc;
            ob[((size_t)b * T_SZ + t) * D_MODEL + h * D_HEAD + d] = f2bf(o[j][r] * g[r]);
        }
    }
}

// ---------------------------------------------------------------------------
// 5) Output projection GEMM (4096 x 2048 x 2048), f32 result.
//    Same double-buffered 32x64 wave tile.
// ---------------------------------------------------------------------------
__global__ __launch_bounds__(256) void gemm_out_kernel(const unsigned short* __restrict__ ab,
                                                       const unsigned short* __restrict__ bt,
                                                       float* __restrict__ out) {
    int lane = threadIdx.x & 31;
    int wave = threadIdx.x >> 5;
    int m0 = blockIdx.y * 256 + wave * 32;
    int n0 = blockIdx.x * 64;
    v8f c[2][4] = {};
    FragBF a[2], b[4], an[2], bn[4];
    a[0] = load_a_frag(ab, D_MODEL, m0,      0, lane);
    a[1] = load_a_frag(ab, D_MODEL, m0 + 16, 0, lane);
    #pragma unroll
    for (int j = 0; j < 4; ++j) b[j] = load_bt_frag(bt, D_MODEL, n0 + j * 16, 0, lane);

    for (int k0 = 0; k0 < D_MODEL; k0 += 32) {
        int kn = k0 + 32;
        if (kn < D_MODEL) {
            an[0] = load_a_frag(ab, D_MODEL, m0,      kn, lane);
            an[1] = load_a_frag(ab, D_MODEL, m0 + 16, kn, lane);
            #pragma unroll
            for (int j = 0; j < 4; ++j)
                bn[j] = load_bt_frag(bt, D_MODEL, n0 + j * 16, kn, lane);
        }
        #pragma unroll
        for (int i = 0; i < 2; ++i)
            #pragma unroll
            for (int j = 0; j < 4; ++j)
                c[i][j] = wmma_bf16(a[i], b[j], c[i][j]);
        a[0] = an[0]; a[1] = an[1];
        #pragma unroll
        for (int j = 0; j < 4; ++j) b[j] = bn[j];
    }
    int half = lane >> 4, nloc = lane & 15;
    #pragma unroll
    for (int i = 0; i < 2; ++i)
        #pragma unroll
        for (int j = 0; j < 4; ++j)
            #pragma unroll
            for (int r = 0; r < 8; ++r)
                out[(size_t)(m0 + i * 16 + r + 8 * half) * D_MODEL + n0 + j * 16 + nloc]
                    = c[i][j][r];
}

// ---------------------------------------------------------------------------
// Host-side launch
// ---------------------------------------------------------------------------
extern "C" void kernel_launch(void* const* d_in, const int* in_sizes, int n_in,
                              void* d_out, int out_size, void* d_ws, size_t ws_size,
                              hipStream_t stream) {
    const float* x        = (const float*)d_in[0];
    const float* w_router = (const float*)d_in[1];
    const float* w_qkv    = (const float*)d_in[2];
    const float* w_out    = (const float*)d_in[3];
    float* out = (float*)d_out;

    char* ws = (char*)d_ws;
    size_t off = 0;
    auto take = [&](size_t bytes) {
        void* p = ws + off;
        off = (off + bytes + 255) & ~(size_t)255;
        return p;
    };
    float*          gate    = (float*)take((size_t)NTOK * H_TOTAL * 4);
    unsigned short* x_bf    = (unsigned short*)take((size_t)NTOK * D_MODEL * 2);
    unsigned short* wqkv_t  = (unsigned short*)take((size_t)N_QKV * D_MODEL * 2);
    unsigned short* wout_t  = (unsigned short*)take((size_t)D_MODEL * D_MODEL * 2);
    unsigned short* q_bf    = (unsigned short*)take((size_t)NTOK * D_MODEL * 2);
    unsigned short* k_bf    = (unsigned short*)take((size_t)NTOK * D_MODEL * 2);
    unsigned short* vt_bf   = (unsigned short*)take((size_t)NTOK * D_MODEL * 2);
    unsigned short* attn_bf = (unsigned short*)take((size_t)NTOK * D_MODEL * 2);
    (void)ws_size; (void)in_sizes; (void)n_in; (void)out_size;

    // 1) router gate
    router_kernel<<<NTOK / 8, 256, 0, stream>>>(x, w_router, gate);
    // 2) casts / transposes
    {
        int n = NTOK * D_MODEL;
        cast_bf16_kernel<<<(n / 4 + 255) / 256, 256, 0, stream>>>(x, x_bf, n);
        transpose_cast_kernel<<<dim3(N_QKV / 32, D_MODEL / 32), 256, 0, stream>>>(
            w_qkv, wqkv_t, D_MODEL, N_QKV);
        transpose_cast_kernel<<<dim3(D_MODEL / 32, D_MODEL / 32), 256, 0, stream>>>(
            w_out, wout_t, D_MODEL, D_MODEL);
    }
    // 3) QKV + RoPE (block tile 256x64)
    qkv_rope_kernel<<<dim3(N_QKV / 64, NTOK / 256), 256, 0, stream>>>(
        x_bf, wqkv_t, q_bf, k_bf, vt_bf);
    // 4) attention (one wave per q-tile per head)
    attn_kernel<<<dim3(T_SZ / 16, B_SZ * H_TOTAL), 32, 0, stream>>>(
        q_bf, k_bf, vt_bf, gate, attn_bf);
    // 5) output projection (block tile 256x64)
    gemm_out_kernel<<<dim3(D_MODEL / 64, NTOK / 256), 256, 0, stream>>>(
        attn_bf, wout_t, out);
}
